// ESRNN_model_22050362097966
// MI455X (gfx1250) — matrix operand order repaired and checked by reading, not compiled, not run on hardware
//
#include <hip/hip_runtime.h>
#include <hip/hip_bf16.h>
#include <stdint.h>

// ---------------- problem constants ----------------
#define BSZ   512      // batch
#define TLEN  256      // train length
#define SIN   12       // input window
#define SOUT  12       // output horizon
#define CCAT  6        // categories
#define HH    128      // hidden
#define G4H   512      // 4*H gate width
#define WOUT  233      // Wn - OUT = (256-12+1) - 12
#define KX0   32       // padded input feature width (18 -> 32)
#define TPAD  236      // 233 padded to multiple of 4
#define TFOLD 59       // 236 / 4
#define MROWS ((size_t)WOUT * BSZ)   // 119296 rows for head GEMMs

typedef __attribute__((ext_vector_type(16))) _Float16 v16h;
typedef __attribute__((ext_vector_type(8)))  float    v8f;

#define WMMA_F16(acc, a, b) \
  (acc) = __builtin_amdgcn_wmma_f32_16x16x32_f16(false, (a), false, (b), (short)0, (acc), false, false)

// fast transcendentals: raw v_rcp_f32, and gfx1250 v_tanh_f32 when clang exposes it
__device__ __forceinline__ float fast_sigmoid(float x) {
  return __builtin_amdgcn_rcpf(1.f + __expf(-x));
}
__device__ __forceinline__ float fast_tanh(float x) {
#if __has_builtin(__builtin_amdgcn_tanhf)
  return __builtin_amdgcn_tanhf(x);
#elif __has_builtin(__builtin_amdgcn_tanh_f32)
  return __builtin_amdgcn_tanh_f32(x);
#else
  // tanh(x) = 1 - 2/(exp(2x)+1)
  return 1.f - 2.f * __builtin_amdgcn_rcpf(__expf(2.f * x) + 1.f);
#endif
}
__device__ __forceinline__ float sigmoid_exact(float x) { return 1.f / (1.f + __expf(-x)); }

// A fragment: 16x32 tile of a row-major [M,K] activation, stride lda (halves).
// Per lane: one contiguous 16-half chunk at k = 16*hf -> compiler emits 2x b128 loads.
__device__ __forceinline__ v16h frag_a(const _Float16* A, int lda) {
  const int lane = threadIdx.x & 31;
  const int m = lane & 15, hf = lane >> 4;
  const _Float16* row = A + (size_t)m * lda;
  v16h a;
#pragma unroll
  for (int p = 0; p < 8; ++p) {
    const int k = 2 * p + 16 * hf;
    a[2 * p]     = row[k];
    a[2 * p + 1] = row[k + 1];
  }
  return a;
}

// B fragment: 32x16 tile where B[k][n] = W[n][k], W row-major [N,K], stride ldw.
__device__ __forceinline__ v16h frag_bT(const _Float16* W, int ldw) {
  const int lane = threadIdx.x & 31;
  const int n = lane & 15, hf = lane >> 4;
  const _Float16* row = W + (size_t)n * ldw + 16 * hf;
  v16h b;
#pragma unroll
  for (int e = 0; e < 16; ++e) b[e] = row[e];
  return b;
}

// ---------------- utility kernels ----------------
__global__ void k_zero_f16(_Float16* d, int n) {
  int i = blockIdx.x * blockDim.x + threadIdx.x;
  if (i < n) d[i] = (_Float16)0.f;
}
__global__ void k_f32_to_f16(const float* s, _Float16* d, int n) {
  int i = blockIdx.x * blockDim.x + threadIdx.x;
  if (i < n) d[i] = (_Float16)s[i];
}
// pad K dimension: src [rows,kin] row-major -> dst [rows,kout] zero-padded
__global__ void k_pad_convert(const float* s, _Float16* d, int rows, int kin, int kout) {
  int i = blockIdx.x * blockDim.x + threadIdx.x;
  if (i >= rows * kout) return;
  int r = i / kout, k = i % kout;
  d[i] = (k < kin) ? (_Float16)s[r * kin + k] : (_Float16)0.f;
}

// ---------------- Holt-Winters scan (1 thread / series) ----------------
__global__ void k_hw(const float* __restrict__ train, const float* __restrict__ lev_sms,
                     const float* __restrict__ seas_sms, const float* __restrict__ init_seas,
                     const int* __restrict__ idxs,
                     float* __restrict__ deseas, float* __restrict__ levels) {
  int b = blockIdx.x * blockDim.x + threadIdx.x;
  if (b >= BSZ) return;
  const int id = idxs[b];
  const float alpha = sigmoid_exact(lev_sms[id]);
  const float gamma = sigmoid_exact(seas_sms[id]);
  float buf[13];  // rolling seasonality buffer == seas_full window
#pragma unroll
  for (int j = 0; j < 12; ++j) buf[j] = init_seas[id * 12 + j];
  buf[12] = buf[0];
  const float y0 = train[(size_t)b * TLEN];
  float lev = y0 / buf[0];
  levels[(size_t)b * TLEN] = lev;
  deseas[(size_t)b * TLEN] = y0 / buf[0];
  for (int t = 1; t < TLEN; ++t) {
    const float y = train[(size_t)b * TLEN + t];
    const float s = buf[0];
    const float nlev = alpha * (y / s) + (1.f - alpha) * lev;
    const float snew = gamma * (y / nlev) + (1.f - gamma) * s;
#pragma unroll
    for (int j = 0; j < 12; ++j) buf[j] = buf[j + 1];
    buf[12] = snew;
    lev = nlev;
    levels[(size_t)b * TLEN + t] = lev;
    deseas[(size_t)b * TLEN + t] = y / buf[0];  // buf[0] == seas_full[t]
  }
}

// ---------------- build x_in f16 [WOUT, B, 32] ----------------
__global__ void k_build_x(const float* __restrict__ deseas, const float* __restrict__ levels,
                          const float* __restrict__ info_cat, _Float16* __restrict__ x) {
  int i = blockIdx.x * blockDim.x + threadIdx.x;
  if (i >= WOUT * BSZ) return;
  const int w = i / BSZ, b = i % BSZ;
  const float invlev = 1.f / levels[(size_t)b * TLEN + (SIN - 1) + w];
  _Float16* xr = x + (size_t)i * KX0;
#pragma unroll
  for (int j = 0; j < SIN; ++j) xr[j] = (_Float16)(deseas[(size_t)b * TLEN + w + j] * invlev);
#pragma unroll
  for (int j = 0; j < CCAT; ++j) xr[SIN + j] = (_Float16)info_cat[b * CCAT + j];
#pragma unroll
  for (int j = SIN + CCAT; j < KX0; ++j) xr[j] = (_Float16)0.f;
}

// ---------------- LSTM layer 0 (dil 1): 32 blocks x 16 batch rows ----------------
// block = 512 threads = 16 waves; wave w owns gate columns [32w, 32w+32)
__global__ __launch_bounds__(512) void k_lstm0(
    const _Float16* __restrict__ xin,   // [WOUT, B, 32]
    const _Float16* __restrict__ Wx,    // [512, 32] f16 (padded Wih0)
    const _Float16* __restrict__ Wh,    // [512, 128] f16
    const float* __restrict__ bias,     // [512]
    _Float16* __restrict__ hout,        // [TPAD, B, 128] (rows >= WOUT pre-zeroed)
    int steps) {
  __shared__ float    gates[16][G4H];   // 32 KB
  __shared__ _Float16 hbuf[16][136];    // hidden state, padded stride
  __shared__ float    cbuf[16][HH];     // cell state
  const int tid  = threadIdx.x;
  const int wave = tid >> 5;
  const int lane = tid & 31;
  const int r0   = blockIdx.x * 16;
  const int em   = tid >> 5;            // elementwise row 0..15
  const int ej   = (tid & 31) * 4;      // elementwise col base (4 contiguous)

  // stage recurrent + input weights as B fragments in registers
  v16h bx[2];
  v16h bh[2][4];
#pragma unroll
  for (int nt = 0; nt < 2; ++nt) {
    const int n0 = wave * 32 + nt * 16;
    bx[nt] = frag_bT(Wx + (size_t)n0 * KX0, KX0);
#pragma unroll
    for (int kt = 0; kt < 4; ++kt)
      bh[nt][kt] = frag_bT(Wh + (size_t)n0 * HH + kt * 32, HH);
  }
  const float bias0 = bias[wave * 32 + (lane & 15)];
  const float bias1 = bias[wave * 32 + 16 + (lane & 15)];

#pragma unroll
  for (int q = 0; q < 4; ++q) {
    hbuf[em][ej + q] = (_Float16)0.f;
    cbuf[em][ej + q] = 0.f;
  }
  __syncthreads();

  for (int t = 0; t < steps; ++t) {
    const _Float16* xt = xin + ((size_t)t * BSZ + r0) * KX0;
    if (t + 1 < steps)
      __builtin_prefetch(xin + ((size_t)(t + 1) * BSZ + r0) * KX0, 0, 0);
    // preload all A fragments so DS/VMEM loads pipeline ahead of the WMMA chain
    v16h ax = frag_a(xt, KX0);
    v16h ah[4];
#pragma unroll
    for (int kt = 0; kt < 4; ++kt) ah[kt] = frag_a(&hbuf[0][kt * 32], 136);

    v8f acc0, acc1;
#pragma unroll
    for (int v = 0; v < 8; ++v) { acc0[v] = bias0; acc1[v] = bias1; }
    WMMA_F16(acc0, ax, bx[0]);
    WMMA_F16(acc1, ax, bx[1]);
#pragma unroll
    for (int kt = 0; kt < 4; ++kt) {
      WMMA_F16(acc0, ah[kt], bh[0][kt]);
      WMMA_F16(acc1, ah[kt], bh[1][kt]);
    }
    const int n = lane & 15, mb = 8 * (lane >> 4);
#pragma unroll
    for (int v = 0; v < 8; ++v) {
      gates[mb + v][wave * 32 + n]      = acc0[v];
      gates[mb + v][wave * 32 + 16 + n] = acc1[v];
    }
    __syncthreads();
    // elementwise LSTM cell: 4 contiguous columns per thread -> vector LDS ops
#pragma unroll
    for (int q = 0; q < 4; ++q) {
      const int j = ej + q;
      const float gi = gates[em][j], gf = gates[em][HH + j];
      const float gg = gates[em][2 * HH + j], go = gates[em][3 * HH + j];
      const float c = fast_sigmoid(gf) * cbuf[em][j] + fast_sigmoid(gi) * fast_tanh(gg);
      const float h = fast_sigmoid(go) * fast_tanh(c);
      cbuf[em][j] = c;
      hbuf[em][j] = (_Float16)h;
      hout[((size_t)t * BSZ + r0 + em) * HH + j] = (_Float16)h;
    }
    __syncthreads();
  }
}

// ---------------- LSTM layer 1 (dil 4, folded): 128 blocks x 16 folded rows ----------------
__global__ __launch_bounds__(512) void k_lstm1(
    const _Float16* __restrict__ h0,    // [TPAD, B, 128]
    const _Float16* __restrict__ Wx,    // [512, 128] f16 (Wih1)
    const _Float16* __restrict__ Wh,    // [512, 128] f16 (Whh1)
    const float* __restrict__ bias,     // [512]
    _Float16* __restrict__ hsum) {      // [WOUT, B, 128] : h1 + h0 (residual)
  __shared__ float    gates[16][G4H];
  __shared__ _Float16 hbuf[16][136];
  __shared__ float    cbuf[16][HH];
  const int tid  = threadIdx.x;
  const int wave = tid >> 5;
  const int lane = tid & 31;
  const int r0    = blockIdx.x * 16;    // folded row base (0..2047)
  const int phase = r0 >> 9;            // tp % 4
  const int b0    = r0 & (BSZ - 1);     // batch base
  const int em    = tid >> 5;
  const int ej    = (tid & 31) * 4;

  v16h bx[2][4];
  v16h bh[2][4];
#pragma unroll
  for (int nt = 0; nt < 2; ++nt) {
    const int n0 = wave * 32 + nt * 16;
#pragma unroll
    for (int kt = 0; kt < 4; ++kt) {
      bx[nt][kt] = frag_bT(Wx + (size_t)n0 * HH + kt * 32, HH);
      bh[nt][kt] = frag_bT(Wh + (size_t)n0 * HH + kt * 32, HH);
    }
  }
  const float bias0 = bias[wave * 32 + (lane & 15)];
  const float bias1 = bias[wave * 32 + 16 + (lane & 15)];

#pragma unroll
  for (int q = 0; q < 4; ++q) {
    hbuf[em][ej + q] = (_Float16)0.f;
    cbuf[em][ej + q] = 0.f;
  }
  __syncthreads();

  for (int u = 0; u < TFOLD; ++u) {
    const int tp = u * 4 + phase;
    const _Float16* xt = h0 + ((size_t)tp * BSZ + b0) * HH;
    if (u + 1 < TFOLD)
      __builtin_prefetch(h0 + ((size_t)(tp + 4) * BSZ + b0) * HH, 0, 0);
    v16h ax[4], ah[4];
#pragma unroll
    for (int kt = 0; kt < 4; ++kt) {
      ax[kt] = frag_a(xt + kt * 32, HH);
      ah[kt] = frag_a(&hbuf[0][kt * 32], 136);
    }
    v8f acc0, acc1;
#pragma unroll
    for (int v = 0; v < 8; ++v) { acc0[v] = bias0; acc1[v] = bias1; }
#pragma unroll
    for (int kt = 0; kt < 4; ++kt) {
      WMMA_F16(acc0, ax[kt], bx[0][kt]);
      WMMA_F16(acc1, ax[kt], bx[1][kt]);
      WMMA_F16(acc0, ah[kt], bh[0][kt]);
      WMMA_F16(acc1, ah[kt], bh[1][kt]);
    }
    const int n = lane & 15, mb = 8 * (lane >> 4);
#pragma unroll
    for (int v = 0; v < 8; ++v) {
      gates[mb + v][wave * 32 + n]      = acc0[v];
      gates[mb + v][wave * 32 + 16 + n] = acc1[v];
    }
    __syncthreads();
#pragma unroll
    for (int q = 0; q < 4; ++q) {
      const int j = ej + q;
      const float gi = gates[em][j], gf = gates[em][HH + j];
      const float gg = gates[em][2 * HH + j], go = gates[em][3 * HH + j];
      const float c = fast_sigmoid(gf) * cbuf[em][j] + fast_sigmoid(gi) * fast_tanh(gg);
      const float h = fast_sigmoid(go) * fast_tanh(c);
      cbuf[em][j] = c;
      hbuf[em][j] = (_Float16)h;
      if (tp < WOUT) {
        const float xv = (float)xt[(size_t)em * HH + j];  // residual = layer input
        hsum[((size_t)tp * BSZ + b0 + em) * HH + j] = (_Float16)(h + xv);
      }
    }
    __syncthreads();
  }
}

// ---------------- head: tanh(hsum @ nlW^T + nl_b) @ scW^T + sc_b ----------------
// one block per 16 rows; 8 waves; wave w -> nl n-tile w; wave 0 does scoring
__global__ __launch_bounds__(256) void k_head(
    const _Float16* __restrict__ hsum,  // [M, 128]
    const _Float16* __restrict__ nlW,   // [128, 128] f16
    const float* __restrict__ nl_b,     // [128]
    const _Float16* __restrict__ scW,   // [16, 128] f16 (rows 12..15 zero)
    const float* __restrict__ sc_b,     // [12]
    float* __restrict__ out) {          // [M, 12]
  __shared__ _Float16 hb[16][136];
  const size_t m0 = (size_t)blockIdx.x * 16;
  const int tid = threadIdx.x, wave = tid >> 5, lane = tid & 31;
  const int n = lane & 15, hf = lane >> 4;

  v8f acc;
  const float bn = nl_b[wave * 16 + n];
#pragma unroll
  for (int v = 0; v < 8; ++v) acc[v] = bn;
  v16h a0[4], bw0[4];
#pragma unroll
  for (int kt = 0; kt < 4; ++kt) {
    a0[kt]  = frag_a(hsum + m0 * HH + kt * 32, HH);
    bw0[kt] = frag_bT(nlW + (size_t)(wave * 16) * HH + kt * 32, HH);
  }
#pragma unroll
  for (int kt = 0; kt < 4; ++kt) WMMA_F16(acc, a0[kt], bw0[kt]);
#pragma unroll
  for (int v = 0; v < 8; ++v) hb[8 * hf + v][wave * 16 + n] = (_Float16)fast_tanh(acc[v]);
  __syncthreads();

  if (wave == 0) {
    v8f a2;
    const float bs = (n < SOUT) ? sc_b[n] : 0.f;
#pragma unroll
    for (int v = 0; v < 8; ++v) a2[v] = bs;
#pragma unroll
    for (int kt = 0; kt < 4; ++kt) {
      v16h a  = frag_a(&hb[0][kt * 32], 136);
      v16h bw = frag_bT(scW + kt * 32, HH);
      WMMA_F16(a2, a, bw);
    }
    if (n < SOUT) {
#pragma unroll
      for (int v = 0; v < 8; ++v)
        out[(m0 + 8 * hf + v) * SOUT + n] = a2[v];
    }
  }
}

// ---------------- launch ----------------
#define CEILDIV(a, b) (((a) + (b) - 1) / (b))

extern "C" void kernel_launch(void* const* d_in, const int* in_sizes, int n_in,
                              void* d_out, int out_size, void* d_ws, size_t ws_size,
                              hipStream_t stream) {
  (void)in_sizes; (void)n_in; (void)out_size; (void)ws_size;
  const float* train    = (const float*)d_in[0];
  const float* info_cat = (const float*)d_in[1];
  const int*   idxs     = (const int*)d_in[2];
  const float* lev_sms  = (const float*)d_in[3];
  const float* seas_sms = (const float*)d_in[4];
  const float* init_seas= (const float*)d_in[5];
  const float* Wih0     = (const float*)d_in[6];
  const float* Whh0     = (const float*)d_in[7];
  const float* b0       = (const float*)d_in[8];
  const float* Wih1     = (const float*)d_in[9];
  const float* Whh1     = (const float*)d_in[10];
  const float* b1       = (const float*)d_in[11];
  const float* nl_W     = (const float*)d_in[12];
  const float* nl_b     = (const float*)d_in[13];
  const float* sc_W     = (const float*)d_in[14];
  const float* sc_b     = (const float*)d_in[15];
  float* out = (float*)d_out;

  char* ws = (char*)d_ws;
  size_t off = 0;
  auto take = [&](size_t bytes) { size_t o = off; off += (bytes + 255) & ~(size_t)255; return o; };
  float*    deseas = (float*)   (ws + take((size_t)BSZ * TLEN * 4));
  float*    levels = (float*)   (ws + take((size_t)BSZ * TLEN * 4));
  _Float16* xf     = (_Float16*)(ws + take((size_t)WOUT * BSZ * KX0 * 2));
  _Float16* h0     = (_Float16*)(ws + take((size_t)TPAD * BSZ * HH * 2));
  _Float16* hsum   = (_Float16*)(ws + take((size_t)WOUT * BSZ * HH * 2));
  _Float16* wih0h  = (_Float16*)(ws + take((size_t)G4H * KX0 * 2));
  _Float16* whh0h  = (_Float16*)(ws + take((size_t)G4H * HH * 2));
  _Float16* wih1h  = (_Float16*)(ws + take((size_t)G4H * HH * 2));
  _Float16* whh1h  = (_Float16*)(ws + take((size_t)G4H * HH * 2));
  _Float16* nlwh   = (_Float16*)(ws + take((size_t)HH * HH * 2));
  _Float16* scwh   = (_Float16*)(ws + take((size_t)16 * HH * 2));

  const int TB = 256;
  // zero: h0 pad rows (WOUT..TPAD-1) and scW pad
  {
    int n = (TPAD - WOUT) * BSZ * HH;
    k_zero_f16<<<CEILDIV(n, TB), TB, 0, stream>>>(h0 + (size_t)WOUT * BSZ * HH, n);
    k_zero_f16<<<CEILDIV(16 * HH, TB), TB, 0, stream>>>(scwh, 16 * HH);
  }
  // weight conversions
  k_pad_convert<<<CEILDIV(G4H * KX0, TB), TB, 0, stream>>>(Wih0, wih0h, G4H, SIN + CCAT, KX0);
  k_f32_to_f16<<<CEILDIV(G4H * HH, TB), TB, 0, stream>>>(Whh0, whh0h, G4H * HH);
  k_f32_to_f16<<<CEILDIV(G4H * HH, TB), TB, 0, stream>>>(Wih1, wih1h, G4H * HH);
  k_f32_to_f16<<<CEILDIV(G4H * HH, TB), TB, 0, stream>>>(Whh1, whh1h, G4H * HH);
  k_f32_to_f16<<<CEILDIV(HH * HH, TB), TB, 0, stream>>>(nl_W, nlwh, HH * HH);
  k_f32_to_f16<<<CEILDIV(SOUT * HH, TB), TB, 0, stream>>>(sc_W, scwh, SOUT * HH);
  // Holt-Winters + windows
  k_hw<<<CEILDIV(BSZ, TB), TB, 0, stream>>>(train, lev_sms, seas_sms, init_seas, idxs,
                                            deseas, levels);
  k_build_x<<<CEILDIV(WOUT * BSZ, TB), TB, 0, stream>>>(deseas, levels, info_cat, xf);
  // recurrent stack
  k_lstm0<<<BSZ / 16, 512, 0, stream>>>(xf, wih0h, whh0h, b0, h0, WOUT);
  k_lstm1<<<(4 * BSZ) / 16, 512, 0, stream>>>(h0, wih1h, whh1h, b1, hsum);
  // head
  k_head<<<(int)(MROWS / 16), 256, 0, stream>>>(hsum, nlwh, nl_b, scwh, sc_b, out);
}
